// CrossAtt_79130477462013
// MI455X (gfx1250) — compile-verified, matrix operand order checked by reading, not run on hardware
//
#include <hip/hip_runtime.h>
#include <hip/hip_bf16.h>

typedef __bf16 bf16_t;
typedef __attribute__((ext_vector_type(16))) __bf16 v16bf;
typedef __attribute__((ext_vector_type(8)))  float  v8f;
typedef unsigned int   u32;
typedef unsigned short u16;

#define DMODEL 768
#define NHEAD  12
#define HDIM   64

// ---------------------------------------------------------------------------
// CDNA5 16-bit A-fragment K index (CDNA5 ISA 7.12.2, 16-bit A 16x32):
// lanes 0-15 hold M=0..15, lanes 16-31 hold M=0..15;
// per lane, elements e=0..7 -> K = half*8 + e, e=8..15 -> K = 16 + half*8 + (e-8)
// For any fixed row the 16 elements are two contiguous 8-element (16B) runs,
// so every fragment load below is 2x ds_load_b128 when the LDS tile is
// K-major for that operand.
__device__ __forceinline__ int a_k_of(int lane, int e) {
    int half = (lane >> 4) & 1;
    return ((e >> 3) << 4) + (half << 3) + (e & 7);
}

__device__ __forceinline__ v8f wmma_bf16(v16bf a, v16bf b, v8f c) {
    return __builtin_amdgcn_wmma_f32_16x16x32_bf16(false, a, false, b,
                                                   (short)0, c, false, false);
}

// ---------------------------------------------------------------------------
__global__ __launch_bounds__(256) void f32_to_bf16_kernel(
    const float* __restrict__ src, bf16_t* __restrict__ dst, size_t n) {
    size_t i = (size_t)blockIdx.x * blockDim.x + threadIdx.x;
    size_t stride = (size_t)gridDim.x * blockDim.x;
    for (; i < n; i += stride) dst[i] = (bf16_t)src[i];
}

// ---------------------------------------------------------------------------
__global__ __launch_bounds__(256) void mask_count_kernel(
    const int* __restrict__ m1, const int* __restrict__ m2,
    int* __restrict__ vl, int Bn, int Lk, int Lq) {
    __shared__ int r1[8], r2[8];
    int b = blockIdx.x;
    int c1 = 0, c2 = 0;
    for (int i = threadIdx.x; i < Lk; i += 256) c1 += (m1[b * Lk + i] != 0);
    for (int i = threadIdx.x; i < Lq; i += 256) c2 += (m2[b * Lq + i] != 0);
#pragma unroll
    for (int off = 16; off; off >>= 1) {
        c1 += __shfl_xor(c1, off, 32);
        c2 += __shfl_xor(c2, off, 32);
    }
    int lane = threadIdx.x & 31, wv = threadIdx.x >> 5;
    if (lane == 0) { r1[wv] = c1; r2[wv] = c2; }
    __syncthreads();
    if (threadIdx.x == 0) {
        int t1 = 0, t2 = 0;
        for (int i = 0; i < 8; ++i) { t1 += r1[i]; t2 += r2[i]; }
        vl[b] = t1;
        vl[Bn + b] = t2;
    }
}

// ---------------------------------------------------------------------------
// bf16 GEMM: C[M,N] = A[M,K] @ B[K,N] (+bias, optional relu), f32 accumulate.
// Block tile 128x128, K step 64. 8 waves in a 4(M) x 2(N) grid; wave tile
// 32x64 (two 16-row M-subtiles x four 16-col N-subtiles -> 16 WMMA/round,
// 24 ds_load_b128/round = 1.5 loads per WMMA). B tile stored K-major in LDS
// (Bst[n][k]) so B-fragment loads are contiguous 16B runs like A.
#define GEMM_BM 128
#define GEMM_BN 128
#define GEMM_BK 64
__global__ __launch_bounds__(256) void gemm_bf16_kernel(
    const bf16_t* __restrict__ A, const bf16_t* __restrict__ Bw,
    float* __restrict__ Cf, bf16_t* __restrict__ Cbf,
    const float* __restrict__ bias, int M, int N, int K, int relu) {
    __shared__ bf16_t As [GEMM_BM][GEMM_BK];   // 128x64, row = 128B (16KB)
    __shared__ bf16_t Bst[GEMM_BN][GEMM_BK];   // transposed [n][k] (16KB)

    const int tid   = threadIdx.x;
    const int lane  = tid & 31;
    const int wave  = tid >> 5;
    const int waveM = wave >> 1;          // 0..3 -> 32 rows each
    const int waveN = wave & 1;           // 0..1 -> 64 cols each
    const int bm    = blockIdx.x * GEMM_BM;
    const int bn    = blockIdx.y * GEMM_BN;
    const int row16 = lane & 15;
    const int half  = (lane >> 4) & 1;

    v8f acc[2][4] = {};

    for (int k0 = 0; k0 < K; k0 += GEMM_BK) {
        {   // A tile: 128x64, each thread 32 bf16 = four 16B copies
            int r = tid >> 1, c = (tid & 1) * 32;
            const uint4* src = (const uint4*)(A + (size_t)(bm + r) * K + k0 + c);
            uint4* dst = (uint4*)&As[r][c];
#pragma unroll
            for (int i = 0; i < 4; ++i) dst[i] = src[i];
        }
        // B tile transposed into LDS: unit = 8 cols x 2 k-rows; pack the two
        // k values of a column into one u32 store (k-pairs are even-aligned).
#pragma unroll
        for (int it = 0; it < 2; ++it) {
            int u  = tid + it * 256;          // 0..511
            int kp = (u & 31) * 2;            // k within tile, even
            int c0 = (u >> 5) * 8;            // 8 columns
            uint4 ra = *(const uint4*)(Bw + (size_t)(k0 + kp)     * N + bn + c0);
            uint4 rb = *(const uint4*)(Bw + (size_t)(k0 + kp + 1) * N + bn + c0);
            const u16* pa = (const u16*)&ra;
            const u16* pb = (const u16*)&rb;
#pragma unroll
            for (int i = 0; i < 8; ++i) {
                u32 v = (u32)pa[i] | ((u32)pb[i] << 16);
                *(u32*)&Bst[c0 + i][kp] = v;
            }
        }
        __syncthreads();

#pragma unroll
        for (int s = 0; s < 2; ++s) {         // two K=32 sub-steps
            v16bf af[2];
#pragma unroll
            for (int mi = 0; mi < 2; ++mi)
#pragma unroll
                for (int e = 0; e < 16; ++e)
                    af[mi][e] = As[waveM * 32 + mi * 16 + row16]
                                  [s * 32 + a_k_of(lane, e)];
#pragma unroll
            for (int nt = 0; nt < 4; ++nt) {
                v16bf bf;
#pragma unroll
                for (int e = 0; e < 16; ++e)
                    bf[e] = Bst[waveN * 64 + nt * 16 + row16]
                               [s * 32 + a_k_of(lane, e)];
                acc[0][nt] = wmma_bf16(af[0], bf, acc[0][nt]);
                acc[1][nt] = wmma_bf16(af[1], bf, acc[1][nt]);
            }
        }
        __syncthreads();
    }

    // Epilogue: C layout (7.12.2): lane holds col n = lane&15; VGPR r -> row
    // m = r + half*8 of the 16x16 tile.
#pragma unroll
    for (int mi = 0; mi < 2; ++mi) {
#pragma unroll
        for (int nt = 0; nt < 4; ++nt) {
            int col = bn + waveN * 64 + nt * 16 + row16;
            float bv = bias ? bias[col] : 0.0f;
#pragma unroll
            for (int r = 0; r < 8; ++r) {
                int m = bm + waveM * 32 + mi * 16 + half * 8 + r;
                float v = acc[mi][nt][r] + bv;
                if (relu) v = v > 0.0f ? v : 0.0f;
                if (Cf)  Cf [(size_t)m * N + col] = v;
                if (Cbf) Cbf[(size_t)m * N + col] = (bf16_t)v;
            }
        }
    }
}

// ---------------------------------------------------------------------------
// Flash attention: block = 4 waves, one (b,h), 64 q rows (16 per wave).
// Ks is [key][dim] (K-contiguous for the QK^T B-fragments); V is stored
// transposed Vst[dim][key] (K-contiguous for the PV B-fragments).
#define ATT_LT 64
__global__ __launch_bounds__(128) void flash_attn_kernel(
    const bf16_t* __restrict__ Q, const bf16_t* __restrict__ Kb,
    const bf16_t* __restrict__ Vb, bf16_t* __restrict__ O,
    const int* __restrict__ vl, int Bn, int Lq, int Lk) {
    __shared__ bf16_t Ks [ATT_LT][HDIM];
    __shared__ bf16_t Vst[HDIM][ATT_LT];
    __shared__ bf16_t Ps[4][16][ATT_LT];   // per-wave P staging, K-contiguous

    const int tid   = threadIdx.x;
    const int lane  = tid & 31;
    const int wave  = tid >> 5;
    const int bh    = blockIdx.y;
    const int b     = bh / NHEAD;
    const int h     = bh % NHEAD;
    const int qbase = blockIdx.x * 64 + wave * 16;
    const int row16 = lane & 15;
    const int half  = (lane >> 4) & 1;
    const int vlen_k = vl[b];
    const int vlen_q = vl[Bn + b];
    const float scale = 0.125f;            // 1/sqrt(64)

    // Q A-fragments (16 rows x 64 dims -> two K=32 fragments); each fragment
    // is two contiguous 16B runs per lane -> global_load_b128 pairs.
    v16bf qf[2];
    {
        const bf16_t* qp =
            Q + (size_t)(b * Lq + qbase + row16) * DMODEL + h * HDIM;
#pragma unroll
        for (int f = 0; f < 2; ++f)
#pragma unroll
            for (int e = 0; e < 16; ++e)
                qf[f][e] = qp[f * 32 + a_k_of(lane, e)];
    }

    v8f oacc[4] = {};
    float mrow[8], lrow[8];
#pragma unroll
    for (int r = 0; r < 8; ++r) { mrow[r] = -3.0e38f; lrow[r] = 0.0f; }

    for (int kt = 0; kt < Lk; kt += ATT_LT) {
        {   // K tile: [key][dim], 16B vector copies (32 bf16 per thread)
            int r = tid >> 1, c0 = (tid & 1) * 32;
            const uint4* ks =
                (const uint4*)(Kb + (size_t)(b * Lk + kt + r) * DMODEL + h * HDIM + c0);
            uint4* kd = (uint4*)&Ks[r][c0];
#pragma unroll
            for (int i = 0; i < 4; ++i) kd[i] = ks[i];
        }
        // V tile transposed: unit = 8 dims x 2 keys, packed u32 stores
#pragma unroll
        for (int it = 0; it < 2; ++it) {
            int u  = tid + it * 128;          // 0..255
            int kp = (u & 31) * 2;            // key pair, even
            int d0 = (u >> 5) * 8;            // 8 dims
            uint4 ra = *(const uint4*)(Vb + (size_t)(b * Lk + kt + kp)     * DMODEL + h * HDIM + d0);
            uint4 rb = *(const uint4*)(Vb + (size_t)(b * Lk + kt + kp + 1) * DMODEL + h * HDIM + d0);
            const u16* pa = (const u16*)&ra;
            const u16* pb = (const u16*)&rb;
#pragma unroll
            for (int i = 0; i < 8; ++i) {
                u32 v = (u32)pa[i] | ((u32)pb[i] << 16);
                *(u32*)&Vst[d0 + i][kp] = v;
            }
        }
        __syncthreads();

        // S = Q K^T over this tile: 4 col sub-tiles of 16 keys
        v8f st[4];
#pragma unroll
        for (int nc = 0; nc < 4; ++nc) {
            v16bf bk0, bk1;
#pragma unroll
            for (int e = 0; e < 16; ++e) {
                int kk = a_k_of(lane, e);
                bk0[e] = Ks[nc * 16 + row16][kk];        // B[k][n] = K[n][k]
                bk1[e] = Ks[nc * 16 + row16][32 + kk];
            }
            v8f z = {};
            z = wmma_bf16(qf[0], bk0, z);
            z = wmma_bf16(qf[1], bk1, z);
            st[nc] = z;
        }

        // scale + dual-axis prefix mask
#pragma unroll
        for (int nc = 0; nc < 4; ++nc) {
            int col = kt + nc * 16 + row16;
#pragma unroll
            for (int r = 0; r < 8; ++r) {
                int qrow = qbase + half * 8 + r;
                float s = st[nc][r] * scale;
                if (col >= vlen_k || qrow >= vlen_q) s = -1.0e6f;
                st[nc][r] = s;
            }
        }

        // online softmax: row max across 4 tiles + 16 lanes of the half-wave
        float corr[8];
#pragma unroll
        for (int r = 0; r < 8; ++r) {
            float m = fmaxf(fmaxf(st[0][r], st[1][r]),
                            fmaxf(st[2][r], st[3][r]));
#pragma unroll
            for (int off = 8; off; off >>= 1)
                m = fmaxf(m, __shfl_xor(m, off, 16));
            float nm = fmaxf(mrow[r], m);
            corr[r] = __expf(mrow[r] - nm);
            mrow[r] = nm;
        }

        // P = exp(S - m): stage to LDS (K-contiguous rows) + row sums
#pragma unroll
        for (int nc = 0; nc < 4; ++nc) {
#pragma unroll
            for (int r = 0; r < 8; ++r) {
                float p = __expf(st[nc][r] - mrow[r]);
                st[nc][r] = p;
                Ps[wave][half * 8 + r][nc * 16 + row16] = (bf16_t)p;
            }
        }
#pragma unroll
        for (int r = 0; r < 8; ++r) {
            float s = st[0][r] + st[1][r] + st[2][r] + st[3][r];
#pragma unroll
            for (int off = 8; off; off >>= 1)
                s += __shfl_xor(s, off, 16);
            lrow[r] = lrow[r] * corr[r] + s;
        }
#pragma unroll
        for (int nd = 0; nd < 4; ++nd)
#pragma unroll
            for (int r = 0; r < 8; ++r)
                oacc[nd][r] *= corr[r];

        // O += P @ V  (wave-private LDS: DS ops are in-order within a wave)
        v16bf pf0, pf1;
#pragma unroll
        for (int e = 0; e < 16; ++e) {
            int kk = a_k_of(lane, e);
            pf0[e] = Ps[wave][row16][kk];
            pf1[e] = Ps[wave][row16][32 + kk];
        }
#pragma unroll
        for (int nd = 0; nd < 4; ++nd) {
            v16bf bv0, bv1;
#pragma unroll
            for (int e = 0; e < 16; ++e) {
                int kk = a_k_of(lane, e);
                bv0[e] = Vst[nd * 16 + row16][kk];       // B[k][n] = V[k][n]
                bv1[e] = Vst[nd * 16 + row16][32 + kk];
            }
            oacc[nd] = wmma_bf16(pf0, bv0, oacc[nd]);
            oacc[nd] = wmma_bf16(pf1, bv1, oacc[nd]);
        }
        __syncthreads();
    }

    // normalize and write head-interleaved [B*Lq, D] bf16
#pragma unroll
    for (int nd = 0; nd < 4; ++nd) {
        int dim = nd * 16 + row16;
#pragma unroll
        for (int r = 0; r < 8; ++r) {
            int qrow = qbase + half * 8 + r;
            float v = oacc[nd][r] / lrow[r];
            O[(size_t)(b * Lq + qrow) * DMODEL + h * HDIM + dim] = (bf16_t)v;
        }
    }
}

// ---------------------------------------------------------------------------
// out = LayerNorm(ffn + mha) * g + b  (f32), one block per token row
__global__ __launch_bounds__(256) void ln_residual_kernel(
    const float* __restrict__ ffn, const float* __restrict__ mha,
    const float* __restrict__ g, const float* __restrict__ be,
    float* __restrict__ out) {
    __shared__ float red[8];
    __shared__ float stats[2];
    const int N = DMODEL;
    size_t row = blockIdx.x;
    const float* fp = ffn + row * N;
    const float* mp = mha + row * N;
    float x[3];
    float s = 0.f;
#pragma unroll
    for (int i = 0; i < 3; ++i) {
        int c = threadIdx.x + 256 * i;
        x[i] = fp[c] + mp[c];
        s += x[i];
    }
#pragma unroll
    for (int off = 16; off; off >>= 1) s += __shfl_xor(s, off, 32);
    int lane = threadIdx.x & 31, wv = threadIdx.x >> 5;
    if (lane == 0) red[wv] = s;
    __syncthreads();
    if (threadIdx.x == 0) {
        float t = 0.f;
        for (int i = 0; i < 8; ++i) t += red[i];
        stats[0] = t / N;
    }
    __syncthreads();
    float mean = stats[0];
    float v = 0.f;
#pragma unroll
    for (int i = 0; i < 3; ++i) { float d = x[i] - mean; v += d * d; }
#pragma unroll
    for (int off = 16; off; off >>= 1) v += __shfl_xor(v, off, 32);
    __syncthreads();
    if (lane == 0) red[wv] = v;
    __syncthreads();
    if (threadIdx.x == 0) {
        float t = 0.f;
        for (int i = 0; i < 8; ++i) t += red[i];
        stats[1] = rsqrtf(t / N + 1e-5f);
    }
    __syncthreads();
    float inv = stats[1];
#pragma unroll
    for (int i = 0; i < 3; ++i) {
        int c = threadIdx.x + 256 * i;
        out[row * N + c] = (x[i] - mean) * inv * g[c] + be[c];
    }
}

// ---------------------------------------------------------------------------
extern "C" void kernel_launch(void* const* d_in, const int* in_sizes, int n_in,
                              void* d_out, int out_size, void* d_ws, size_t ws_size,
                              hipStream_t stream) {
    (void)in_sizes; (void)n_in; (void)out_size; (void)ws_size;
    const int B = 8, Lq = 1024, Lk = 1024;
    const size_t AT = (size_t)B * Lq * DMODEL;   // 6,291,456 token elements
    const size_t WE = (size_t)DMODEL * DMODEL;

    const float* queries = (const float*)d_in[0];
    const float* keysf   = (const float*)d_in[1];
    const float* valuesf = (const float*)d_in[2];
    const int*   mask1   = (const int*)d_in[3];
    const int*   mask2   = (const int*)d_in[4];
    const float* Ww[6] = { (const float*)d_in[5],  (const float*)d_in[6],
                           (const float*)d_in[7],  (const float*)d_in[8],
                           (const float*)d_in[9],  (const float*)d_in[11] };
    const float* d1_b = (const float*)d_in[10];
    const float* d2_b = (const float*)d_in[12];
    const float* ln_g = (const float*)d_in[13];
    const float* ln_b = (const float*)d_in[14];
    float* out = (float*)d_out;

    // ---- workspace layout (256B aligned) ----
    char* ws = (char*)d_ws;
    size_t off = 0;
    auto take = [&](size_t bytes) {
        size_t o = off;
        off = (off + bytes + 255) & ~(size_t)255;
        return o;
    };
    int*    vl     = (int*)   (ws + take(64));
    bf16_t* wbf    = (bf16_t*)(ws + take(6 * WE * 2));
    bf16_t* xq_bf  = (bf16_t*)(ws + take(AT * 2));
    bf16_t* xk_bf  = (bf16_t*)(ws + take(AT * 2));
    bf16_t* xv_bf  = (bf16_t*)(ws + take(AT * 2));
    bf16_t* q_bf   = (bf16_t*)(ws + take(AT * 2));
    bf16_t* k_bf   = (bf16_t*)(ws + take(AT * 2));
    bf16_t* v_bf   = (bf16_t*)(ws + take(AT * 2));
    float*  mha_f  = (float*) (ws + take(AT * 4));
    float*  ffn2_f = (float*) (ws + take(AT * 4));
    // safe aliases (producer/consumer ordering on the single stream):
    bf16_t* attn_bf = xq_bf;   // xq dead after Q projection
    bf16_t* mha_bf  = q_bf;    // q dead after attention
    bf16_t* ffn1_bf = k_bf;    // k dead after attention

    // ---- 1. mask valid-length counts ----
    mask_count_kernel<<<B, 256, 0, stream>>>(mask1, mask2, vl, B, Lk, Lq);

    // ---- 2. f32 -> bf16 conversions ----
    f32_to_bf16_kernel<<<4096, 256, 0, stream>>>(queries, xq_bf, AT);
    f32_to_bf16_kernel<<<4096, 256, 0, stream>>>(keysf,   xk_bf, AT);
    f32_to_bf16_kernel<<<4096, 256, 0, stream>>>(valuesf, xv_bf, AT);
    for (int i = 0; i < 6; ++i)
        f32_to_bf16_kernel<<<1024, 256, 0, stream>>>(Ww[i], wbf + i * WE, WE);

    const int M = B * Lq;                       // 8192
    dim3 ggrid(M / GEMM_BM, DMODEL / GEMM_BN);  // (64, 6)

    // ---- 3. Q/K/V projections ----
    gemm_bf16_kernel<<<ggrid, 256, 0, stream>>>(xq_bf, wbf + 0 * WE, nullptr,
                                                q_bf, nullptr, M, DMODEL, DMODEL, 0);
    gemm_bf16_kernel<<<ggrid, 256, 0, stream>>>(xk_bf, wbf + 1 * WE, nullptr,
                                                k_bf, nullptr, M, DMODEL, DMODEL, 0);
    gemm_bf16_kernel<<<ggrid, 256, 0, stream>>>(xv_bf, wbf + 2 * WE, nullptr,
                                                v_bf, nullptr, M, DMODEL, DMODEL, 0);

    // ---- 4. flash attention ----
    dim3 agrid(Lq / 64, B * NHEAD);             // (16, 96)
    flash_attn_kernel<<<agrid, 128, 0, stream>>>(q_bf, k_bf, v_bf, attn_bf,
                                                 vl, B, Lq, Lk);

    // ---- 5. output projection (mha in f32 for residual, bf16 for d1) ----
    gemm_bf16_kernel<<<ggrid, 256, 0, stream>>>(attn_bf, wbf + 3 * WE, mha_f,
                                                mha_bf, nullptr, M, DMODEL, DMODEL, 0);

    // ---- 6. FFN: relu(mha @ d1 + b1) @ d2 + b2 ----
    gemm_bf16_kernel<<<ggrid, 256, 0, stream>>>(mha_bf, wbf + 4 * WE, nullptr,
                                                ffn1_bf, d1_b, M, DMODEL, DMODEL, 1);
    gemm_bf16_kernel<<<ggrid, 256, 0, stream>>>(ffn1_bf, wbf + 5 * WE, ffn2_f,
                                                nullptr, d2_b, M, DMODEL, DMODEL, 0);

    // ---- 7. residual + layernorm ----
    ln_residual_kernel<<<M, 256, 0, stream>>>(ffn2_f, mha_f, ln_g, ln_b, out);
}
